// CopyLayer_26422638805523
// MI455X (gfx1250) — matrix operand order repaired and checked by reading, not compiled, not run on hardware
//
#include <hip/hip_runtime.h>
#include <stdint.h>

#define B_ 8
#define N_ 2048
#define V_ 256
#define H_ 1024
#define M_ (B_ * N_)           // 16384 rows
#define CHUNK 128
#define NCHUNK (N_ / CHUNK)    // 16
#define BM 32                  // M-tile of fused MLP kernel (64KB LDS for Y tile)

typedef __attribute__((ext_vector_type(16))) __bf16        v16bf;
typedef __attribute__((ext_vector_type(8)))  float         v8f;
typedef __attribute__((ext_vector_type(16))) unsigned short v16u;
typedef __attribute__((ext_vector_type(8)))  unsigned short v8u;

static __device__ __forceinline__ unsigned short f32_to_bf16(float f) {
    uint32_t u = __float_as_uint(f);
    uint32_t r = u + 0x7FFFu + ((u >> 16) & 1u);   // round-to-nearest-even
    return (unsigned short)(r >> 16);
}

// ---------------------------------------------------------------------------
// 1) BOS-row LayerNorms (attn & mlp flavors) + bosdir[b] = lnA . qk_direction
// ---------------------------------------------------------------------------
__global__ void k_prep_bos(const float* __restrict__ h,
                           const float* __restrict__ ga, const float* __restrict__ bea,
                           const float* __restrict__ gm, const float* __restrict__ bem,
                           const float* __restrict__ qdir,
                           float* __restrict__ lnA, float* __restrict__ lnM,
                           float* __restrict__ bosdir) {
    __shared__ float red[V_];
    int b = blockIdx.x, f = threadIdx.x;
    float x = h[(size_t)b * N_ * V_ + f];
    red[f] = x; __syncthreads();
    for (int s = V_ / 2; s > 0; s >>= 1) { if (f < s) red[f] += red[f + s]; __syncthreads(); }
    float mean = red[0] * (1.0f / V_); __syncthreads();
    float dx = x - mean;
    red[f] = dx * dx; __syncthreads();
    for (int s = V_ / 2; s > 0; s >>= 1) { if (f < s) red[f] += red[f + s]; __syncthreads(); }
    float rstd = rsqrtf(red[0] * (1.0f / V_) + 1e-5f);
    float la = dx * rstd * ga[f] + bea[f];
    float lm = dx * rstd * gm[f] + bem[f];
    lnA[b * V_ + f] = la;
    lnM[b * V_ + f] = lm;
    __syncthreads();
    red[f] = la * qdir[f]; __syncthreads();
    for (int s = V_ / 2; s > 0; s >>= 1) { if (f < s) red[f] += red[f + s]; __syncthreads(); }
    if (f == 0) bosdir[b] = red[0];
}

// ---------------------------------------------------------------------------
// 2) v_bos = wo_w @ wv_bos   (256x256 GEMV, tiny)
// ---------------------------------------------------------------------------
__global__ void k_vbos(const float* __restrict__ wo, const float* __restrict__ wvb,
                       float* __restrict__ vbos) {
    __shared__ float sh[V_];
    int i = threadIdx.x;
    sh[i] = wvb[i];
    __syncthreads();
    float s = 0.f;
    const float* row = wo + (size_t)i * V_;
    for (int j = 0; j < V_; ++j) s += row[j] * sh[j];
    vbos[i] = s;
}

// ---------------------------------------------------------------------------
// 3) bf16 weight conversions
// ---------------------------------------------------------------------------
__global__ void k_convw(const float* __restrict__ w, unsigned short* __restrict__ wb) {
    int idx = blockIdx.x * 256 + threadIdx.x;
    wb[idx] = f32_to_bf16(w[idx]);
}

// ---------------------------------------------------------------------------
// 4) Per-row softmax coefficients from the sparse QK structure.
//    One wave32 per row; shuffle reductions for the two 256-dim dots.
// ---------------------------------------------------------------------------
__global__ void k_qk(const float* __restrict__ h, const float* __restrict__ lnA,
                     const float* __restrict__ qb, const float* __restrict__ qp,
                     const float* __restrict__ bosdir,
                     float* __restrict__ cv0, float* __restrict__ cpv,
                     float* __restrict__ csm) {
    int tid = threadIdx.x;
    int wave = tid >> 5, lane = tid & 31;
    int row = blockIdx.x * 8 + wave;              // 16384 rows total
    int b = row >> 11, r = row & (N_ - 1);
    const float* a = (r == 0) ? (lnA + b * V_) : (h + (size_t)row * V_);
    float sb = 0.f, sp = 0.f;
    for (int i = 0; i < 8; ++i) {
        int f = lane + 32 * i;
        float x = a[f];
        sb += x * qb[f];
        sp += x * qp[f];
    }
    for (int off = 16; off > 0; off >>= 1) {
        sb += __shfl_down(sb, off);
        sp += __shfl_down(sp, off);
    }
    if (lane == 0) {
        float c0, cp, cs;
        if (r == 0) {                 // row 0: softmax is a one-hot on col 0
            c0 = 1.f; cp = 0.f; cs = 0.f;
        } else {
            float col0 = sb * bosdir[b];
            float dd   = sp;
            if (r == 1) {             // col 0 holds col0+d; diagonal holds 0
                float l0 = col0 + dd;
                float m  = fmaxf(l0, 0.f);
                float e0 = expf(l0 - m), ez = expf(-m);
                float Z  = e0 + ez;
                c0 = e0 / Z; cp = 0.f; cs = ez / Z;   // S[1] == val[1]
            } else {
                float m  = fmaxf(fmaxf(col0, dd), 0.f);
                float e0 = expf(col0 - m), ed = expf(dd - m), ez = expf(-m);
                float Z  = e0 + ed + (float)(r - 1) * ez;
                c0 = e0 / Z; cp = (ed - ez) / Z; cs = ez / Z;
            }
        }
        cv0[row] = c0; cpv[row] = cp; csm[row] = cs;
    }
}

// ---------------------------------------------------------------------------
// 5a) chunk partial sums of val[c] = h[c]*wv   (c >= 1)
// ---------------------------------------------------------------------------
__global__ void k_csum(const float* __restrict__ h, const float* __restrict__ wv,
                       float* __restrict__ cs) {
    int bc = blockIdx.x;                    // b*NCHUNK + c
    int b = bc >> 4, c = bc & (NCHUNK - 1);
    int f = threadIdx.x;
    float wvf = wv[f];
    int r0 = c * CHUNK;
    const float* base = h + ((size_t)b * N_ + r0) * V_ + f;
    float s = 0.f;
    for (int i = 0; i < CHUNK; ++i) {
        if (r0 + i >= 1) s += base[(size_t)i * V_] * wvf;
    }
    cs[(size_t)bc * V_ + f] = s;
}

// 5b) in-place exclusive prefix over the 16 chunks, per (b, f)
__global__ void k_cpre(float* __restrict__ cs) {
    int b = blockIdx.x, f = threadIdx.x;
    float run = 0.f;
    for (int c = 0; c < NCHUNK; ++c) {
        size_t idx = ((size_t)b * NCHUNK + c) * V_ + f;
        float t = cs[idx];
        cs[idx] = run;
        run += t;
    }
}

// 5c) per-chunk scan: writes attention result into d_out AND emits the bf16
//     mlp_input (BOS row swapped with lnM) as a byproduct of the same h pass.
__global__ void k_scan(const float* __restrict__ h, const float* __restrict__ wv,
                       const float* __restrict__ vbos,
                       const float* __restrict__ cv0, const float* __restrict__ cpv,
                       const float* __restrict__ csm, const float* __restrict__ cs,
                       const float* __restrict__ lnM,
                       float* __restrict__ out, unsigned short* __restrict__ xb) {
    __shared__ float s0[CHUNK], s1[CHUNK], s2[CHUNK];
    int bc = blockIdx.x;
    int b = bc >> 4, c = bc & (NCHUNK - 1);
    int f = threadIdx.x;
    int r0 = c * CHUNK;
    int rowbase = b * N_ + r0;
    if (f < CHUNK) {
        s0[f] = cv0[rowbase + f];
        s1[f] = cpv[rowbase + f];
        s2[f] = csm[rowbase + f];
    }
    __syncthreads();
    float wvf = wv[f], vb = vbos[f];
    float S = cs[(size_t)bc * V_ + f];
    float vprev = (r0 >= 2) ? h[((size_t)b * N_ + r0 - 1) * V_ + f] * wvf : 0.f;
    const float* hb = h + (size_t)rowbase * V_ + f;
    float* ob = out + (size_t)rowbase * V_ + f;
    unsigned short* xbp = xb + (size_t)rowbase * V_ + f;
    for (int i = 0; i < CHUNK; ++i) {
        int r = r0 + i;
        float hx, vr = 0.f;
        if (r >= 1) {
            hx = hb[(size_t)i * V_];
            vr = hx * wvf;
            S += vr;
        } else {
            hx = lnM[b * V_ + f];   // BOS row of mlp_input is LayerNorm'd
        }
        ob[(size_t)i * V_] = s0[i] * vb + s1[i] * vprev + s2[i] * S;
        xbp[(size_t)i * V_] = f32_to_bf16(hx);
        vprev = vr;
    }
}

// ---------------------------------------------------------------------------
// 6) Fused MLP: OUT += relu(X @ W1^T) @ W2^T  via bf16 WMMA, Y staged in LDS.
//    Fragment layouts (A and B use contiguous k = 16*half + s, not AMD's table order; A read as two
//    adjacent 16B K-chunks, B as one 32B K-chunk per lane; C/D: VGPR r -> row r+8*half).
//    LDS Y tile is 32x1024 bf16 (64KB) with XOR bank swizzle on the 16B chunk
//    index (chunk ^ (row & 7)) so phase-2 row-parallel reads spread over banks
//    while every access stays 16B-aligned (ds_load_b128).
// ---------------------------------------------------------------------------
__global__ void __launch_bounds__(256) k_mlp(
    const unsigned short* __restrict__ xb, const unsigned short* __restrict__ w1b,
    const unsigned short* __restrict__ w2b, float* __restrict__ out) {
    __shared__ unsigned short ylds[BM * 1024];    // 64 KB
    int tid = threadIdx.x;
    int wave = tid >> 5, lane = tid & 31;
    int lrow = lane & 15, lhalf = lane >> 4;
    int m0 = blockIdx.x * BM;
    int mt = wave >> 2;          // waves 0-3 -> m-tile 0, waves 4-7 -> m-tile 1
    int wq = wave & 3;

    // ---- Phase 1: Y = relu(X @ W1^T) -> LDS (bf16, swizzled) ----
    // Hoist the 8 A fragments (shared by this wave's 16 N-tiles).
    const unsigned short* arow = xb + (size_t)(m0 + mt * 16 + lrow) * V_;
    v16u afrag[8];
    for (int ks = 0; ks < 8; ++ks) {
        v8u alo = *(const v8u*)(arow + ks * 32 + 16 * lhalf);
        v8u ahi = *(const v8u*)(arow + ks * 32 + 16 * lhalf + 8);
        afrag[ks] = __builtin_shufflevector(alo, ahi,
            0, 1, 2, 3, 4, 5, 6, 7, 8, 9, 10, 11, 12, 13, 14, 15);
    }
    for (int i = 0; i < 16; ++i) {
        int nt = wq * 16 + i;                         // 0..63 N-tiles
        v8f acc = {};
        const unsigned short* brow = w1b + (size_t)(nt * 16 + lrow) * V_;
        for (int ks = 0; ks < 8; ++ks) {
            v16u b16 = *(const v16u*)(brow + ks * 32 + 16 * lhalf);
            acc = __builtin_amdgcn_wmma_f32_16x16x32_bf16(
                false, __builtin_bit_cast(v16bf, afrag[ks]),
                false, __builtin_bit_cast(v16bf, b16),
                (short)0, acc, false, false);
        }
        int n = nt * 16 + lrow;
        int cch = n >> 3;                             // 16B chunk index in row
        for (int r = 0; r < 8; ++r) {
            int ml = mt * 16 + r + 8 * lhalf;
            int addr = ml * 1024 + (((cch ^ (ml & 7)) << 3) | (n & 7));
            float v = acc[r];
            ylds[addr] = f32_to_bf16(v > 0.f ? v : 0.f);
        }
    }
    __syncthreads();

    // ---- Phase 2: OUT += Y @ W2^T ----
    // A fragment depends only on ks -> load once, feed 4 output N-tiles.
    int ml = mt * 16 + lrow;
    int sw = ml & 7;
    const unsigned short* yrow = &ylds[ml * 1024];
    v8f acc0 = {}, acc1 = {}, acc2 = {}, acc3 = {};
    const unsigned short* b0 = w2b + (size_t)((wq * 4 + 0) * 16 + lrow) * H_;
    const unsigned short* b1 = w2b + (size_t)((wq * 4 + 1) * 16 + lrow) * H_;
    const unsigned short* b2 = w2b + (size_t)((wq * 4 + 2) * 16 + lrow) * H_;
    const unsigned short* b3 = w2b + (size_t)((wq * 4 + 3) * 16 + lrow) * H_;
    for (int ks = 0; ks < 32; ++ks) {
        int clo = ks * 4 + 2 * lhalf;                 // lo/hi 16B chunk indices
        int chi = clo + 1;
        v8u alo = *(const v8u*)(yrow + ((clo ^ sw) << 3));
        v8u ahi = *(const v8u*)(yrow + ((chi ^ sw) << 3));
        v16u a16 = __builtin_shufflevector(alo, ahi,
            0, 1, 2, 3, 4, 5, 6, 7, 8, 9, 10, 11, 12, 13, 14, 15);
        v16bf av = __builtin_bit_cast(v16bf, a16);
        int ko = ks * 32 + 16 * lhalf;
        v16u w0 = *(const v16u*)(b0 + ko);
        v16u w1v = *(const v16u*)(b1 + ko);
        v16u w2v = *(const v16u*)(b2 + ko);
        v16u w3 = *(const v16u*)(b3 + ko);
        acc0 = __builtin_amdgcn_wmma_f32_16x16x32_bf16(
            false, av, false, __builtin_bit_cast(v16bf, w0), (short)0, acc0, false, false);
        acc1 = __builtin_amdgcn_wmma_f32_16x16x32_bf16(
            false, av, false, __builtin_bit_cast(v16bf, w1v), (short)0, acc1, false, false);
        acc2 = __builtin_amdgcn_wmma_f32_16x16x32_bf16(
            false, av, false, __builtin_bit_cast(v16bf, w2v), (short)0, acc2, false, false);
        acc3 = __builtin_amdgcn_wmma_f32_16x16x32_bf16(
            false, av, false, __builtin_bit_cast(v16bf, w3), (short)0, acc3, false, false);
    }
    for (int j = 0; j < 4; ++j) {
        v8f acc = (j == 0) ? acc0 : (j == 1) ? acc1 : (j == 2) ? acc2 : acc3;
        int nt = wq * 4 + j;
        int n = nt * 16 + lrow;
        for (int r = 0; r < 8; ++r) {
            int m = m0 + mt * 16 + r + 8 * lhalf;
            size_t idx = (size_t)m * V_ + n;
            out[idx] = acc[r] + out[idx];             // attn result already there
        }
    }
}

// ---------------------------------------------------------------------------
extern "C" void kernel_launch(void* const* d_in, const int* in_sizes, int n_in,
                              void* d_out, int out_size, void* d_ws, size_t ws_size,
                              hipStream_t stream) {
    const float* h        = (const float*)d_in[0];
    // d_in[1] = mask_one, d_in[2] = mask_zero: structure is exploited analytically
    const float* ln_attn_g = (const float*)d_in[3];
    const float* ln_attn_b = (const float*)d_in[4];
    const float* ln_mlp_g  = (const float*)d_in[5];
    const float* ln_mlp_b  = (const float*)d_in[6];
    const float* wv        = (const float*)d_in[7];
    const float* wv_bos    = (const float*)d_in[8];
    const float* wo_w      = (const float*)d_in[9];
    const float* qk_bos    = (const float*)d_in[10];
    const float* qk_prev   = (const float*)d_in[11];
    const float* qk_dir    = (const float*)d_in[12];
    const float* w1        = (const float*)d_in[13];
    const float* w2        = (const float*)d_in[14];
    float* out = (float*)d_out;

    char* ws = (char*)d_ws;
    size_t off = 0;
    auto alloc = [&](size_t bytes) -> void* {
        void* p = ws + off;
        off += (bytes + 255) & ~(size_t)255;
        return p;
    };
    float* lnA    = (float*)alloc(B_ * V_ * sizeof(float));
    float* lnM    = (float*)alloc(B_ * V_ * sizeof(float));
    float* bosdir = (float*)alloc(B_ * sizeof(float));
    float* vbos   = (float*)alloc(V_ * sizeof(float));
    float* cv0    = (float*)alloc(M_ * sizeof(float));
    float* cpv    = (float*)alloc(M_ * sizeof(float));
    float* csm    = (float*)alloc(M_ * sizeof(float));
    float* cs     = (float*)alloc((size_t)B_ * NCHUNK * V_ * sizeof(float));
    unsigned short* xbuf = (unsigned short*)alloc((size_t)M_ * V_ * sizeof(unsigned short));
    unsigned short* w1b  = (unsigned short*)alloc((size_t)H_ * V_ * sizeof(unsigned short));
    unsigned short* w2b  = (unsigned short*)alloc((size_t)V_ * H_ * sizeof(unsigned short));

    k_prep_bos<<<B_, V_, 0, stream>>>(h, ln_attn_g, ln_attn_b, ln_mlp_g, ln_mlp_b,
                                      qk_dir, lnA, lnM, bosdir);
    k_vbos<<<1, V_, 0, stream>>>(wo_w, wv_bos, vbos);
    k_convw<<<(H_ * V_) / 256, 256, 0, stream>>>(w1, w1b);
    k_convw<<<(V_ * H_) / 256, 256, 0, stream>>>(w2, w2b);
    k_qk<<<M_ / 8, 256, 0, stream>>>(h, lnA, qk_bos, qk_prev, bosdir, cv0, cpv, csm);
    k_csum<<<B_ * NCHUNK, V_, 0, stream>>>(h, wv, cs);
    k_cpre<<<B_, V_, 0, stream>>>(cs);
    k_scan<<<B_ * NCHUNK, V_, 0, stream>>>(h, wv, vbos, cv0, cpv, csm, cs, lnM,
                                           out, xbuf);
    k_mlp<<<M_ / BM, 256, 0, stream>>>(xbuf, w1b, w2b, out);
}
